// _MulHeadAttnModule_61718680044431
// MI455X (gfx1250) — matrix operand order repaired and checked by reading, not compiled, not run on hardware
//
#include <hip/hip_runtime.h>
#include <stdint.h>

// Problem constants (from reference)
#define B_    4
#define T_    1024
#define D_    512
#define H_    8
#define DH_   64
#define L_    2047
#define LPAD_ 2048
#define M_    (B_ * T_)   // 4096 rows of x

typedef unsigned short ushort_t;
typedef __attribute__((ext_vector_type(16))) __bf16        v16bf;
typedef __attribute__((ext_vector_type(8)))  float         v8f;
typedef __attribute__((ext_vector_type(4)))  unsigned int  u32x4;

// Fragment for WMMA A/B operands: 8 VGPRs = 16 bf16 per lane.
// e[2r], e[2r+1] sit in VGPR r low/high halves (ISA 7.12.2 packing).
union FragAB {
    v16bf    v;
    ushort_t e[16];
    u32x4    q[2];
};

__device__ __forceinline__ ushort_t f2bf(float f) {
    uint32_t u = __float_as_uint(f);
    uint32_t r = u + 0x7FFFu + ((u >> 16) & 1u);   // round-to-nearest-even
    return (ushort_t)(r >> 16);
}

__device__ __forceinline__ float redmax16(float x) {
#pragma unroll
    for (int m = 1; m < 16; m <<= 1) x = fmaxf(x, __shfl_xor(x, m, 32));
    return x;
}
__device__ __forceinline__ float redsum16(float x) {
#pragma unroll
    for (int m = 1; m < 16; m <<= 1) x += __shfl_xor(x, m, 32);
    return x;
}

// ---------------------------------------------------------------------------
// f32 -> bf16 row-major copy with zero tail padding (used for x and pos_emb)
// ---------------------------------------------------------------------------
__global__ void k_f32_to_bf16(const float* __restrict__ src,
                              ushort_t* __restrict__ dst,
                              int n_src, int n_total) {
    for (int i = blockIdx.x * blockDim.x + threadIdx.x; i < n_total;
         i += gridDim.x * blockDim.x)
        dst[i] = (i < n_src) ? f2bf(src[i]) : (ushort_t)0;
}

// f32 W[K][N] -> bf16 Wt[N][K]  (K-major operand for WMMA B fragments)
__global__ void k_transpose_f32_to_bf16(const float* __restrict__ W,
                                        ushort_t* __restrict__ Wt,
                                        int K, int N) {
    int i = blockIdx.x * blockDim.x + threadIdx.x;
    if (i >= K * N) return;
    int n = i / K, k = i % K;
    Wt[i] = f2bf(W[k * N + n]);
}

// bf16 V [B*T, H*DH] -> bf16 Vt [B, H, DH, T]  (t-contiguous for P@V B frags)
__global__ void k_transpose_v(const ushort_t* __restrict__ vb,
                              ushort_t* __restrict__ vt) {
    int i = blockIdx.x * blockDim.x + threadIdx.x;
    if (i >= B_ * H_ * DH_ * T_) return;
    int t = i % T_;
    int r = i / T_;
    int d = r % DH_;
    int h = (r / DH_) % H_;
    int b = r / (DH_ * H_);
    vt[i] = vb[(b * T_ + t) * D_ + h * DH_ + d];
}

// ---------------------------------------------------------------------------
// GEMM: C[16x64 tile] = A[M,K]bf16 @ Bt[N,K]bf16^T, fp32 accumulate.
// Optional fused epilogues: bias + add0 -> bf16 out0, bias + add1 -> bf16
// out1 (q projection writes q+u and q+v in one pass), bias -> f32 outf.
// One wave per 16x64 tile; all operands loaded as coalesced b128, no LDS.
// ---------------------------------------------------------------------------
__global__ __launch_bounds__(128)
void k_gemm_bf16_16x64(const ushort_t* __restrict__ A,
                       const ushort_t* __restrict__ Bt,
                       int Mtiles, int Ntiles64, int K,
                       const float* __restrict__ bias,
                       const float* __restrict__ add0, ushort_t* __restrict__ out0,
                       const float* __restrict__ add1, ushort_t* __restrict__ out1,
                       float* __restrict__ outf, int ldc) {
    int lane = threadIdx.x & 31;
    int wave = threadIdx.x >> 5;
    int gw   = blockIdx.x * 4 + wave;
    if (gw >= Mtiles * Ntiles64) return;
    int m0 = (gw / Ntiles64) * 16;
    int n0 = (gw % Ntiles64) * 64;
    int lr   = lane & 15;
    int kb8  = (lane < 16) ? 0 : 8;    // A-fragment K base (ISA 16-bit A layout)
    int kb16 = (lane < 16) ? 0 : 16;   // unused; B fragments use kb8 like A

    v8f cacc[4] = {};
    const ushort_t* arow = A + (size_t)(m0 + lr) * K;
    for (int k0 = 0; k0 < K; k0 += 32) {
        FragAB a;
        a.q[0] = *(const u32x4*)(arow + k0 + kb8);
        a.q[1] = *(const u32x4*)(arow + k0 + 16 + kb8);
#pragma unroll
        for (int nt = 0; nt < 4; ++nt) {
            FragAB b;
            const ushort_t* brow = Bt + (size_t)(n0 + nt * 16 + lr) * K + k0 + kb8;
            b.q[0] = *(const u32x4*)(brow);
            b.q[1] = *(const u32x4*)(brow + 16);
            cacc[nt] = __builtin_amdgcn_wmma_f32_16x16x32_bf16(
                false, a.v, false, b.v, (short)0, cacc[nt], false, false);
        }
    }
#pragma unroll
    for (int nt = 0; nt < 4; ++nt) {
        int ncol   = n0 + nt * 16 + lr;
        float badd = bias ? bias[ncol] : 0.f;
        float a0   = add0 ? add0[ncol] : 0.f;
        float a1   = add1 ? add1[ncol] : 0.f;
#pragma unroll
        for (int r = 0; r < 8; ++r) {
            int mrow  = m0 + ((lane < 16) ? r : r + 8);
            float val = cacc[nt][r] + badd;
            size_t idx = (size_t)mrow * ldc + ncol;
            if (out0) out0[idx] = f2bf(val + a0);
            if (out1) out1[idx] = f2bf(val + a1);
            if (outf) outf[idx] = val;
        }
    }
}

// ---------------------------------------------------------------------------
// Fused flash attention with Transformer-XL relative shift.
// One wave handles one (b, h, 16-row q-tile). j advances by 32 per iter:
//  - AC tile per 16-j subtile: 2 WMMAs vs K fragments loaded straight from HBM
//  - BD: tmp[16x32] = (q+v) @ r[window]^T (4 WMMAs) -> LDS -> diagonal gather
//    tmp[m][n-m+15]  (== bd[q, j-q+T-1], i.e. the rel_shift)
//  - combined online softmax over both subtiles, P -> LDS bf16
//  - P[16x32] @ V[32x64]: 4 WMMAs vs t-contiguous Vt fragments
// ---------------------------------------------------------------------------
__global__ __launch_bounds__(128)
void k_attn(const ushort_t* __restrict__ qu, const ushort_t* __restrict__ qv,
            const ushort_t* __restrict__ kbuf, const ushort_t* __restrict__ rbuf,
            const ushort_t* __restrict__ vt, ushort_t* __restrict__ ao) {
    __shared__ float    tmp_s[4][16 * 32];
    __shared__ ushort_t p_s[4][16 * 32];

    int lane = threadIdx.x & 31;
    int wave = threadIdx.x >> 5;
    int gw   = blockIdx.x * 4 + wave;          // 0 .. B*H*(T/16)-1 = 2047
    int q0   = (gw & 63) * 16;
    int h    = (gw >> 6) & 7;
    int b    = gw >> 9;
    int lr   = lane & 15;
    int kb8  = (lane < 16) ? 0 : 8;
    int kb16 = (lane < 16) ? 0 : 16;
    float*    tmp = tmp_s[wave];
    ushort_t* pl  = p_s[wave];

    // Loop-invariant A fragments: (q+u) and (q+v), two K=32 chunks over DH=64.
    FragAB au[2], av[2];
    {
        const ushort_t* ru = qu + (size_t)(b * T_ + q0 + lr) * D_ + h * DH_;
        const ushort_t* rv = qv + (size_t)(b * T_ + q0 + lr) * D_ + h * DH_;
#pragma unroll
        for (int c = 0; c < 2; ++c) {
            au[c].q[0] = *(const u32x4*)(ru + 32 * c + kb8);
            au[c].q[1] = *(const u32x4*)(ru + 32 * c + 16 + kb8);
            av[c].q[0] = *(const u32x4*)(rv + 32 * c + kb8);
            av[c].q[1] = *(const u32x4*)(rv + 32 * c + 16 + kb8);
        }
    }

    v8f o[4] = {};
    float rm[8], rl[8];
#pragma unroll
    for (int r = 0; r < 8; ++r) { rm[r] = -1e30f; rl[r] = 0.f; }

    for (int j0 = 0; j0 < T_; j0 += 32) {
        v8f s[2];
#pragma unroll
        for (int sub = 0; sub < 2; ++sub) {
            int js = j0 + sub * 16;
            // content term AC
            v8f acc = {};
#pragma unroll
            for (int c = 0; c < 2; ++c) {
                FragAB bk;
                const ushort_t* kr =
                    kbuf + (size_t)(b * T_ + js + lr) * D_ + h * DH_ + 32 * c + kb8;
                bk.q[0] = *(const u32x4*)(kr);
                bk.q[1] = *(const u32x4*)(kr + 16);
                acc = __builtin_amdgcn_wmma_f32_16x16x32_bf16(
                    false, au[c].v, false, bk.v, (short)0, acc, false, false);
            }
            // position term BD over shifted 32-row r window (max l = 2047 pad row = 0)
            int lw = js - q0 + T_ - 16;
#pragma unroll
            for (int nh = 0; nh < 2; ++nh) {
                v8f ct = {};
#pragma unroll
                for (int c = 0; c < 2; ++c) {
                    FragAB br;
                    const ushort_t* rr =
                        rbuf + (size_t)(lw + nh * 16 + lr) * D_ + h * DH_ + 32 * c + kb8;
                    br.q[0] = *(const u32x4*)(rr);
                    br.q[1] = *(const u32x4*)(rr + 16);
                    ct = __builtin_amdgcn_wmma_f32_16x16x32_bf16(
                        false, av[c].v, false, br.v, (short)0, ct, false, false);
                }
#pragma unroll
                for (int r = 0; r < 8; ++r) {
                    int mrow = (lane < 16) ? r : r + 8;
                    tmp[mrow * 32 + nh * 16 + lr] = ct[r];
                }
            }
            // rel_shift diagonal gather: bd[m][n] = tmp[m][n - m + 15]
#pragma unroll
            for (int r = 0; r < 8; ++r) {
                int mrow = (lane < 16) ? r : r + 8;
                float bd = tmp[mrow * 32 + (lr - mrow + 15)];
                s[sub][r] = (acc[r] + bd) * 0.125f;   // 1/sqrt(DH)
            }
        }
        // combined online softmax across the two 16-j subtiles
#pragma unroll
        for (int r = 0; r < 8; ++r) {
            float mx    = redmax16(fmaxf(s[0][r], s[1][r]));
            float mnew  = fmaxf(rm[r], mx);
            float scale = __expf(rm[r] - mnew);
            rm[r]       = mnew;
            float pa = __expf(s[0][r] - mnew);
            float pb = __expf(s[1][r] - mnew);
            rl[r] = rl[r] * scale + redsum16(pa + pb);
#pragma unroll
            for (int nt = 0; nt < 4; ++nt) o[nt][r] *= scale;
            int mrow = (lane < 16) ? r : r + 8;
            pl[mrow * 32 + lr]      = f2bf(pa);
            pl[mrow * 32 + 16 + lr] = f2bf(pb);
        }
        // rebuild P as an A fragment (K = 32 = this iteration's j span)
        FragAB ap;
#pragma unroll
        for (int i = 0; i < 8; ++i) {
            ap.e[i]     = pl[lr * 32 + kb8 + i];
            ap.e[8 + i] = pl[lr * 32 + 16 + kb8 + i];
        }
        // O += P @ V (Vt is [B,H,DH,T]: t-contiguous -> clean B fragments)
#pragma unroll
        for (int nt = 0; nt < 4; ++nt) {
            FragAB bv;
            const ushort_t* vr =
                vt + (size_t)((b * H_ + h) * DH_ + nt * 16 + lr) * T_ + j0 + kb8;
            bv.q[0] = *(const u32x4*)(vr);
            bv.q[1] = *(const u32x4*)(vr + 16);
            o[nt] = __builtin_amdgcn_wmma_f32_16x16x32_bf16(
                false, ap.v, false, bv.v, (short)0, o[nt], false, false);
        }
    }
    // epilogue: normalize and store head-merged [B,T,D] bf16
#pragma unroll
    for (int r = 0; r < 8; ++r) {
        int mrow  = (lane < 16) ? r : r + 8;
        float inv = 1.f / rl[r];
#pragma unroll
        for (int nt = 0; nt < 4; ++nt)
            ao[(size_t)(b * T_ + q0 + mrow) * D_ + h * DH_ + nt * 16 + lr] =
                f2bf(o[nt][r] * inv);
    }
}

// ---------------------------------------------------------------------------
extern "C" void kernel_launch(void* const* d_in, const int* in_sizes, int n_in,
                              void* d_out, int out_size, void* d_ws, size_t ws_size,
                              hipStream_t stream) {
    (void)in_sizes; (void)n_in; (void)out_size; (void)ws_size;
    const float* x   = (const float*)d_in[0];
    const float* pe  = (const float*)d_in[1];
    const float* Wq  = (const float*)d_in[2];
    const float* bq  = (const float*)d_in[3];
    const float* Wk  = (const float*)d_in[4];
    const float* bk  = (const float*)d_in[5];
    const float* Wv  = (const float*)d_in[6];
    const float* bv  = (const float*)d_in[7];
    const float* Wr  = (const float*)d_in[8];
    const float* Wo  = (const float*)d_in[9];
    const float* bo  = (const float*)d_in[10];
    const float* u   = (const float*)d_in[11];  // [H,DH] == flat [D]
    const float* v   = (const float*)d_in[12];
    float* out = (float*)d_out;

    char* ws = (char*)d_ws;
    size_t off = 0;
    auto alloc = [&](size_t bytes) -> void* {
        void* p = ws + off;
        off += (bytes + 255) & ~(size_t)255;
        return p;
    };
    ushort_t* xb   = (ushort_t*)alloc((size_t)M_ * D_ * 2);
    ushort_t* peb  = (ushort_t*)alloc((size_t)LPAD_ * D_ * 2);
    ushort_t* Wqt  = (ushort_t*)alloc((size_t)D_ * D_ * 2);
    ushort_t* Wkt  = (ushort_t*)alloc((size_t)D_ * D_ * 2);
    ushort_t* Wvt  = (ushort_t*)alloc((size_t)D_ * D_ * 2);
    ushort_t* Wrt  = (ushort_t*)alloc((size_t)D_ * D_ * 2);
    ushort_t* Wot  = (ushort_t*)alloc((size_t)D_ * D_ * 2);
    ushort_t* qub  = (ushort_t*)alloc((size_t)M_ * D_ * 2);
    ushort_t* qvb  = (ushort_t*)alloc((size_t)M_ * D_ * 2);
    ushort_t* kbuf = (ushort_t*)alloc((size_t)M_ * D_ * 2);
    ushort_t* vbuf = (ushort_t*)alloc((size_t)M_ * D_ * 2);
    ushort_t* rbuf = (ushort_t*)alloc((size_t)LPAD_ * D_ * 2);
    ushort_t* vtb  = (ushort_t*)alloc((size_t)M_ * D_ * 2);
    ushort_t* aob  = (ushort_t*)alloc((size_t)M_ * D_ * 2);

    // 1) convert / transpose operands to bf16
    k_f32_to_bf16<<<(M_ * D_) / 256, 256, 0, stream>>>(x, xb, M_ * D_, M_ * D_);
    k_f32_to_bf16<<<(LPAD_ * D_) / 256, 256, 0, stream>>>(pe, peb, L_ * D_, LPAD_ * D_);
    k_transpose_f32_to_bf16<<<(D_ * D_) / 256, 256, 0, stream>>>(Wq, Wqt, D_, D_);
    k_transpose_f32_to_bf16<<<(D_ * D_) / 256, 256, 0, stream>>>(Wk, Wkt, D_, D_);
    k_transpose_f32_to_bf16<<<(D_ * D_) / 256, 256, 0, stream>>>(Wv, Wvt, D_, D_);
    k_transpose_f32_to_bf16<<<(D_ * D_) / 256, 256, 0, stream>>>(Wr, Wrt, D_, D_);
    k_transpose_f32_to_bf16<<<(D_ * D_) / 256, 256, 0, stream>>>(Wo, Wot, D_, D_);

    auto gemm = [&](const ushort_t* A, int Mtiles, const ushort_t* Bt,
                    const float* bias, const float* add0, ushort_t* o0,
                    const float* add1, ushort_t* o1, float* of) {
        int tiles  = Mtiles * (D_ / 64);
        int blocks = (tiles + 3) / 4;
        k_gemm_bf16_16x64<<<blocks, 128, 0, stream>>>(
            A, Bt, Mtiles, D_ / 64, D_, bias, add0, o0, add1, o1, of, D_);
    };

    // 2) projections (q-GEMM fuses both +u and +v head-bias variants)
    gemm(xb, M_ / 16, Wqt, bq, u, qub, v, qvb, nullptr);
    gemm(xb, M_ / 16, Wkt, bk, nullptr, kbuf, nullptr, nullptr, nullptr);
    gemm(xb, M_ / 16, Wvt, bv, nullptr, vbuf, nullptr, nullptr, nullptr);
    gemm(peb, LPAD_ / 16, Wrt, nullptr, nullptr, rbuf, nullptr, nullptr, nullptr);

    // 3) V -> [B,H,DH,T] for P@V B fragments
    k_transpose_v<<<(M_ * D_) / 256, 256, 0, stream>>>(vbuf, vtb);

    // 4) fused rel-shift flash attention: B*H*(T/16) = 2048 waves
    k_attn<<<512, 128, 0, stream>>>(qub, qvb, kbuf, rbuf, vtb, aob);

    // 5) output projection, fp32 result
    gemm(aob, M_ / 16, Wot, bo, nullptr, nullptr, nullptr, nullptr, out);
}
